// LightGCNWithKG_68788196212816
// MI455X (gfx1250) — matrix operand order repaired and checked by reading, not compiled, not run on hardware
//
#include <hip/hip_runtime.h>
#include <hip/hip_bf16.h>

// ---------------------------------------------------------------------------
// LightGCN + KG fusion for MI455X (gfx1250, wave32).
//  * Node state kept as bf16 (`xh`) — it is only ever consumed by the bf16
//    WMMA GEMMs, so storing f32 and re-converting each layer was pure waste.
//  * Conv weights pre-swizzled into WMMA B-fragment order (contiguous 32B per
//    lane per fragment) -> conv GEMM inner loop is pure b128 loads + v_wmma.
//  * Edge propagation: one wave/edge, coalesced 256B row gather, native
//    global_atomic_add_f32 into L2-resident xnew (51MB << 192MB L2).
//  * Elementwise passes fused: bias+acc+bf16-store+xnew-rezero in one sweep;
//    final /(L+1) folded into last layer's epilogue.
// Assumes EMB == 64 (reference), power-of-two D.
// ---------------------------------------------------------------------------

typedef __attribute__((ext_vector_type(16))) __bf16 v16bf;
typedef __attribute__((ext_vector_type(8)))  __bf16 v8bf;
typedef __attribute__((ext_vector_type(8)))  float  v8f;

#define WMMA_BF16(a, b, c) \
  __builtin_amdgcn_wmma_f32_16x16x32_bf16(false, (a), false, (b), (short)0, (c), false, false)

// ---------------- KG fusion GEMM: xh[U+m,:] = bf16(item_emb_w[m,:] + feat[m,:] @ fusion_w^T + fusion_b)
__global__ void fusion_gemm_kernel(const float* __restrict__ feat,        // [I, K]
                                   const float* __restrict__ item_emb_w,  // [I, 64]
                                   const float* __restrict__ fusion_w,    // [64, K]
                                   const float* __restrict__ fusion_b,    // [64]
                                   __bf16* __restrict__ xh,               // [N, 64] bf16 state
                                   float*  __restrict__ acc,              // [N, 64] f32 sum
                                   int num_items, int K, int num_users) {
  const int lane  = threadIdx.x & 31;
  const int wave  = blockIdx.x * (blockDim.x >> 5) + (threadIdx.x >> 5);
  const int m0    = wave * 16;
  if (m0 >= num_items) return;               // wave-uniform: EXEC all-ones for WMMA
  const int ncol  = lane & 15;
  const int khalf = lane >> 4;

  int mrow = m0 + ncol;                      // A-fragment row for this lane
  if (mrow >= num_items) mrow = num_items - 1;

  v8f c[4];
  #pragma unroll
  for (int t = 0; t < 4; ++t) c[t] = (v8f){0.f,0.f,0.f,0.f,0.f,0.f,0.f,0.f};

  for (int kb = 0; kb < K; kb += 32) {
    // A 16x32 bf16 fragment: elems 0..7 <- K = kb+khalf*8+j ; elems 8..15 <- +16
    v16bf a;
    const float* ap = feat + (size_t)mrow * K + kb + khalf * 8;
    #pragma unroll
    for (int j = 0; j < 8; ++j) { a[j] = (__bf16)ap[j]; a[8 + j] = (__bf16)ap[16 + j]; }
    // B 32x16: B[k,n] = fusion_w[n, k] -> two runs of 8 floats per lane (+0..7, +16..23)
    #pragma unroll
    for (int t = 0; t < 4; ++t) {
      const float* bp = fusion_w + (size_t)(t * 16 + ncol) * K + kb + khalf * 8;
      v16bf b;
      #pragma unroll
      for (int j = 0; j < 8; ++j) { b[j] = (__bf16)bp[j]; b[8 + j] = (__bf16)bp[16 + j]; }
      c[t] = WMMA_BF16(a, b, c[t]);
    }
  }

  const int gr0 = m0 + khalf * 8;            // D-fragment: elem r -> row gr0+r, col ncol
  #pragma unroll
  for (int t = 0; t < 4; ++t) {
    #pragma unroll
    for (int r = 0; r < 8; ++r) {
      const int m = gr0 + r;
      if (m < num_items) {
        const int nc = t * 16 + ncol;
        const float v = c[t][r] + item_emb_w[(size_t)m * 64 + nc] + fusion_b[nc];
        const size_t o = ((size_t)(num_users + m)) * 64 + nc;
        xh[o] = (__bf16)v;
        acc[o] = v;
      }
    }
  }
}

// ---------------- Pre-swizzle conv weights into B-fragment order.
// whs[l][kc][t][lane][j] = W_l[kc*32 + (lane/16)*8 + (j&7) + (j>>3)*16, t*16 + (lane%15...&15)]
__global__ void pack_w_kernel(const float* __restrict__ cw, __bf16* __restrict__ whs, int total) {
  int i = blockIdx.x * blockDim.x + threadIdx.x;
  if (i >= total) return;
  const int j    = i & 15;
  const int lane = (i >> 4) & 31;
  const int t    = (i >> 9) & 3;
  const int kc   = (i >> 11) & 1;
  const int l    = i >> 12;
  const int k    = kc * 32 + (lane >> 4) * 8 + (j & 7) + (j >> 3) * 16;
  const int n    = t * 16 + (lane & 15);
  whs[i] = (__bf16)cw[((size_t)l * 64 + k) * 64 + n];
}

// ---------------- Per-layer GEMM: xl = xh @ W_l  (bf16 in, f32 out)
__global__ void conv_gemm_kernel(const __bf16* __restrict__ xh,   // [N, 64] bf16
                                 const __bf16* __restrict__ whs,  // per-layer swizzled [2][4][32][16]
                                 float* __restrict__ xl,          // [N, 64] f32
                                 int N) {
  const int lane  = threadIdx.x & 31;
  const int wave  = blockIdx.x * (blockDim.x >> 5) + (threadIdx.x >> 5);
  const int m0    = wave * 16;
  if (m0 >= N) return;                       // wave-uniform
  const int ncol  = lane & 15;
  const int khalf = lane >> 4;

  int mrow = m0 + ncol;
  if (mrow >= N) mrow = N - 1;

  v8f c[4];
  #pragma unroll
  for (int t = 0; t < 4; ++t) c[t] = (v8f){0.f,0.f,0.f,0.f,0.f,0.f,0.f,0.f};

  #pragma unroll
  for (int kc = 0; kc < 2; ++kc) {
    const __bf16* ap = xh + (size_t)mrow * 64 + kc * 32 + khalf * 8;
    const v8bf lo = *(const v8bf*)ap;              // K = kc*32 + khalf*8 + 0..7
    const v8bf hi = *(const v8bf*)(ap + 16);       // K = +16
    const v16bf a = __builtin_shufflevector(lo, hi, 0,1,2,3,4,5,6,7,8,9,10,11,12,13,14,15);
    #pragma unroll
    for (int t = 0; t < 4; ++t) {
      const v16bf b = *(const v16bf*)(whs + (size_t)(((kc * 4 + t) * 32 + lane) * 16));
      c[t] = WMMA_BF16(a, b, c[t]);
    }
  }

  const int gr0 = m0 + khalf * 8;
  #pragma unroll
  for (int t = 0; t < 4; ++t) {
    #pragma unroll
    for (int r = 0; r < 8; ++r) {
      const int m = gr0 + r;
      if (m < N) xl[(size_t)m * 64 + (t * 16 + ncol)] = c[t][r];
    }
  }
}

// ---------------- Elementwise / edge kernels
__global__ void copy_users_kernel(const float* __restrict__ uw, __bf16* __restrict__ xh,
                                  float* __restrict__ acc, size_t n) {
  size_t i = (size_t)blockIdx.x * blockDim.x + threadIdx.x;
  if (i < n) { float v = uw[i]; xh[i] = (__bf16)v; acc[i] = v; }
}

__global__ void zero_kernel(float* __restrict__ p, size_t n) {
  size_t i = (size_t)blockIdx.x * blockDim.x + threadIdx.x;
  if (i < n) p[i] = 0.f;
}

__global__ void degree_kernel(const int* __restrict__ col, float* __restrict__ deg, int E) {
  int i = blockIdx.x * blockDim.x + threadIdx.x;
  if (i < E) unsafeAtomicAdd(&deg[col[i]], 1.0f);   // global_atomic_add_f32 @ L2; exact (E < 2^24)
}

__global__ void dinv_kernel(float* __restrict__ deg, int N) {
  int i = blockIdx.x * blockDim.x + threadIdx.x;
  if (i < N) { float d = deg[i]; deg[i] = (d > 0.f) ? rsqrtf(d) : 0.f; }
}

__global__ void norm_kernel(const int* __restrict__ row, const int* __restrict__ col,
                            const float* __restrict__ dinv, float* __restrict__ norm, int E) {
  int i = blockIdx.x * blockDim.x + threadIdx.x;
  if (i < E) norm[i] = dinv[row[i]] * dinv[col[i]];
}

// One wave per edge; each lane moves 2 floats (float2 gather + 2 f32 atomics).
__global__ void scatter_kernel(const int* __restrict__ row, const int* __restrict__ col,
                               const float* __restrict__ norm, const float* __restrict__ xl,
                               float* __restrict__ xnew, int E) {
  const long long tid = (long long)blockIdx.x * blockDim.x + threadIdx.x;
  const long long e = tid >> 5;
  if (e >= E) return;
  const int lane = threadIdx.x & 31;
  const int j = lane * 2;
  const int r = row[e];
  const int c = col[e];
  const float nm = norm[e];
  const float2 v = *(const float2*)(xl + (size_t)r * 64 + j);
  if (lane == 0 && e + 8 < (long long)E)      // warm the gather stream (global_prefetch_b8)
    __builtin_prefetch(xl + (size_t)row[e + 8] * 64, 0, 0);
  unsafeAtomicAdd(xnew + (size_t)c * 64 + j,     nm * v.x);
  unsafeAtomicAdd(xnew + (size_t)c * 64 + j + 1, nm * v.y);
}

// Fused: v = xnew + b ; xh = bf16(v) ; acc += v ; xnew = 0 (ready for next scatter)
__global__ void bias_acc_kernel(float* __restrict__ xnew, const float* __restrict__ b,
                                __bf16* __restrict__ xh, float* __restrict__ acc,
                                size_t total, int dmask) {
  size_t i = (size_t)blockIdx.x * blockDim.x + threadIdx.x;
  if (i < total) {
    float v = xnew[i] + b[i & (size_t)dmask];
    xnew[i] = 0.f;
    xh[i] = (__bf16)v;
    acc[i] += v;
  }
}

// Last layer: fold the /(L+1) output directly; x/xh/acc no longer needed.
__global__ void bias_acc_final_kernel(const float* __restrict__ xnew, const float* __restrict__ b,
                                      const float* __restrict__ acc, float* __restrict__ out,
                                      size_t total, int dmask, float s) {
  size_t i = (size_t)blockIdx.x * blockDim.x + threadIdx.x;
  if (i < total) {
    float v = xnew[i] + b[i & (size_t)dmask];
    out[i] = (acc[i] + v) * s;
  }
}

// ---------------------------------------------------------------------------
extern "C" void kernel_launch(void* const* d_in, const int* in_sizes, int n_in,
                              void* d_out, int out_size, void* d_ws, size_t ws_size,
                              hipStream_t stream) {
  const int*   eidx = (const int*)d_in[0];          // [2, E] (int32 per harness contract)
  const float* feat = (const float*)d_in[1];        // [I, K]
  const float* uw   = (const float*)d_in[2];        // [U, D]
  const float* iw   = (const float*)d_in[3];        // [I, D]
  const float* fw   = (const float*)d_in[4];        // [D, K]
  const float* fb   = (const float*)d_in[5];        // [D]
  const float* cw   = (const float*)d_in[6];        // [L, D, D]
  const float* cb   = (const float*)d_in[7];        // [L, D]

  const int E = in_sizes[0] / 2;
  const int D = in_sizes[5];                        // 64
  const int K = in_sizes[4] / D;                    // 768
  const int U = in_sizes[2] / D;
  const int I = in_sizes[3] / D;
  const int L = in_sizes[7] / D;
  const int N = U + I;
  const int* row = eidx;
  const int* col = eidx + E;

  // Workspace partition. f32 first, then bf16 (all offsets stay 32B aligned).
  const size_t nd = (size_t)N * D;
  float* p = (float*)d_ws;
  float* acc  = p; p += nd;
  float* xl   = p; p += nd;
  float* xnew = p; p += nd;
  float* nrm  = p; p += (size_t)E;
  float* deg  = p; p += (size_t)N;                  // becomes dinv in-place
  __bf16* xh  = (__bf16*)p;
  __bf16* whs = xh + nd;                            // L * 2*4*32*16 bf16

  const size_t ud = (size_t)U * D;
  const int blk = 256;
  const int wpack = L * 4096;                       // packed weight elements
  auto gs = [](size_t n, int b) { return (unsigned)((n + b - 1) / b); };

  // Stage 0: init deg/xnew, users -> xh/acc, pack weights, fused item GEMM.
  zero_kernel<<<gs((size_t)N, blk), blk, 0, stream>>>(deg, (size_t)N);
  zero_kernel<<<gs(nd, blk), blk, 0, stream>>>(xnew, nd);
  copy_users_kernel<<<gs(ud, blk), blk, 0, stream>>>(uw, xh, acc, ud);
  pack_w_kernel<<<gs((size_t)wpack, blk), blk, 0, stream>>>(cw, whs, wpack);
  {
    const size_t waves = (size_t)(I + 15) / 16;
    fusion_gemm_kernel<<<gs(waves * 32, blk), blk, 0, stream>>>(feat, iw, fw, fb, xh, acc, I, K, U);
  }

  // Stage 1: degree -> dinv -> per-edge norm
  degree_kernel<<<gs((size_t)E, blk), blk, 0, stream>>>(col, deg, E);
  dinv_kernel<<<gs((size_t)N, blk), blk, 0, stream>>>(deg, N);
  norm_kernel<<<gs((size_t)E, blk), blk, 0, stream>>>(row, col, deg, nrm, E);

  // Stage 2: L propagation layers
  const size_t conv_waves = (size_t)(N + 15) / 16;
  for (int l = 0; l < L; ++l) {
    conv_gemm_kernel<<<gs(conv_waves * 32, blk), blk, 0, stream>>>(xh, whs + (size_t)l * 4096, xl, N);
    scatter_kernel<<<gs((size_t)E * 32, blk), blk, 0, stream>>>(row, col, nrm, xl, xnew, E);
    if (l < L - 1) {
      bias_acc_kernel<<<gs(nd, blk), blk, 0, stream>>>(xnew, cb + (size_t)l * D, xh, acc, nd, D - 1);
    } else {
      bias_acc_final_kernel<<<gs(nd, blk), blk, 0, stream>>>(xnew, cb + (size_t)l * D, acc,
                                                             (float*)d_out, nd, D - 1,
                                                             1.0f / (float)(L + 1));
    }
  }
}
